// GraphMessagePassing_25924422598773
// MI455X (gfx1250) — matrix operand order repaired and checked by reading, not compiled, not run on hardware
//
#include <hip/hip_runtime.h>
#include <stdint.h>

#define NN 50000
#define EE 600000
#define DD 128

typedef __attribute__((ext_vector_type(16))) __bf16    v16bf;
typedef __attribute__((ext_vector_type(8)))  float     v8f;
typedef __attribute__((ext_vector_type(8)))  uint32_t  v8u;

// ---------------- workspace layout (bytes) ----------------
#define OFF_AGGR  0
#define OFF_XBF   (NN * DD * 4)
#define OFF_WFRAG (OFF_XBF + NN * DD * 2)

// fragment table bases (fragment = 16x16 N-tile x 32 K-chunk, 512 bf16, ct-major)
#define FB_EW0 0     // edge layer0, K=288 (x_i|x_j|geo|pad): 9*8 = 72
#define FB_EW1 72    // 32
#define FB_EW2 104   // 32
#define FB_NW0 136   // 64
#define FB_NW1 200   // 32
#define FB_NW2 232   // 32
#define NFRAG  264

__device__ __forceinline__ unsigned short f2bf(float f) {
  union { float f; uint32_t u; } c; c.f = f;
  uint32_t u = c.u;
  u += 0x7fffu + ((u >> 16) & 1u);   // round to nearest even
  return (unsigned short)(u >> 16);
}

// A fragment (16x32 bf16) from row-major LDS tile.
__device__ __forceinline__ v16bf load_a_lds(const unsigned short* rowp, int colbase) {
  uint4 lo = *(const uint4*)(rowp + colbase);
  uint4 hi = *(const uint4*)(rowp + colbase + 16);
  v8u t;
  t[0] = lo.x; t[1] = lo.y; t[2] = lo.z; t[3] = lo.w;
  t[4] = hi.x; t[5] = hi.y; t[6] = hi.z; t[7] = hi.w;
  return __builtin_bit_cast(v16bf, t);
}

// B fragment pre-swizzled in workspace: 32 lanes x 16 bf16 contiguous (1 KB).
__device__ __forceinline__ v16bf load_b_ws(const unsigned short* wf, int frag, int lane) {
  return *(const v16bf*)(wf + (size_t)frag * 512 + lane * 16);
}

#define WMMA_BF16(a, b, c) \
  __builtin_amdgcn_wmma_f32_16x16x32_bf16(false, (a), false, (b), (short)0, (c), false, false)

__device__ __forceinline__ void init_acc2(v8f acc0[8], v8f acc1[8], const float* bias, int ln) {
#pragma unroll
  for (int nt = 0; nt < 8; nt++) {
    float b = bias[nt * 16 + ln];
#pragma unroll
    for (int r = 0; r < 8; r++) { acc0[nt][r] = b; acc1[nt][r] = b; }
  }
}

// M=32 layer: two A tiles share every B fragment; B/A prefetched one step ahead.
template <int NCT>
__device__ __forceinline__ void mlp_layer2(v8f acc0[8], v8f acc1[8],
                                           const unsigned short* arow0,
                                           const unsigned short* arow1, int acol0,
                                           const unsigned short* wf, int fb,
                                           int half, int lane) {
  v16bf a0 = load_a_lds(arow0, acol0 + half * 8);
  v16bf a1 = load_a_lds(arow1, acol0 + half * 8);
  v16bf b  = load_b_ws(wf, fb, lane);
#pragma unroll
  for (int i = 0; i < NCT * 8; i++) {
    const int ct = i >> 3, nt = i & 7;
    v16bf bn = b, a0n = a0, a1n = a1;
    if (i + 1 < NCT * 8) bn = load_b_ws(wf, fb + i + 1, lane);
    if (nt == 7 && ct + 1 < NCT) {
      a0n = load_a_lds(arow0, acol0 + (ct + 1) * 32 + half * 8);
      a1n = load_a_lds(arow1, acol0 + (ct + 1) * 32 + half * 8);
    }
    acc0[nt] = WMMA_BF16(a0, b, acc0[nt]);
    acc1[nt] = WMMA_BF16(a1, b, acc1[nt]);
    b = bn; a0 = a0n; a1 = a1n;
  }
}

// ReLU + bf16 store of a 32x128 wave tile into LDS.
__device__ __forceinline__ void store_act_relu2(v8f acc0[8], v8f acc1[8],
                                                unsigned short* wact, int pitch,
                                                int coloff, int half, int ln) {
#pragma unroll
  for (int nt = 0; nt < 8; nt++) {
    int n = nt * 16 + ln;
#pragma unroll
    for (int r = 0; r < 8; r++) {
      wact[(half * 8 + r) * pitch + coloff + n]      = f2bf(fmaxf(acc0[nt][r], 0.0f));
      wact[(16 + half * 8 + r) * pitch + coloff + n] = f2bf(fmaxf(acc1[nt][r], 0.0f));
    }
  }
}

// ---------------- prep kernels ----------------
__global__ void k_zero(uint4* p, int n) {
  int t = blockIdx.x * blockDim.x + threadIdx.x;
  if (t < n) { uint4 z; z.x = z.y = z.z = z.w = 0u; p[t] = z; }
}

__global__ void k_xbf(const float* __restrict__ x, unsigned short* __restrict__ xbf, int n4) {
  int t = blockIdx.x * blockDim.x + threadIdx.x;
  if (t < n4) {
    float4 v = ((const float4*)x)[t];
    uint2 r;
    r.x = (uint32_t)f2bf(v.x) | ((uint32_t)f2bf(v.y) << 16);
    r.y = (uint32_t)f2bf(v.z) | ((uint32_t)f2bf(v.w) << 16);
    ((uint2*)xbf)[t] = r;
  }
}

__global__ void k_prep(const float* __restrict__ ew0, const float* __restrict__ ew1,
                       const float* __restrict__ ew2, const float* __restrict__ nw0,
                       const float* __restrict__ nw1, const float* __restrict__ nw2,
                       unsigned short* __restrict__ wfrag) {
  int tid = blockIdx.x * blockDim.x + threadIdx.x;
  if (tid >= NFRAG * 512) return;
  int f = tid >> 9;
  int e = tid & 511;
  int lane = e >> 4;
  int tt = e & 15;
  int which, lf;
  if      (f < FB_EW1) { which = 0; lf = f; }
  else if (f < FB_EW2) { which = 1; lf = f - FB_EW1; }
  else if (f < FB_NW0) { which = 2; lf = f - FB_EW2; }
  else if (f < FB_NW1) { which = 3; lf = f - FB_NW0; }
  else if (f < FB_NW2) { which = 4; lf = f - FB_NW1; }
  else                 { which = 5; lf = f - FB_NW2; }
  int ct = lf >> 3, nt = lf & 7;
  int k = ct * 32 + ((lane < 16) ? 0 : 8) + (tt & 7) + ((tt >> 3) << 4);   // B layout K index
  int n = nt * 16 + (lane & 15);                   // B layout N column
  float val = 0.0f;
  switch (which) {
    case 0:  // K 0..255 -> ew0 row k+4 (x_i|x_j); 256..259 -> rows 0..3 (d,norm); pad 0
      if (k < 256)      val = ew0[(k + 4) * DD + n];
      else if (k < 260) val = ew0[(k - 256) * DD + n];
      break;
    case 1: val = ew1[k * DD + n]; break;
    case 2: val = ew2[k * DD + n]; break;
    case 3: val = nw0[k * DD + n]; break;
    case 4: val = nw1[k * DD + n]; break;
    case 5: val = nw2[k * DD + n]; break;
  }
  wfrag[(size_t)f * 512 + e] = f2bf(val);
}

// ---------------- edge MLP + scatter ----------------
// 64 threads = 2 waves, each wave owns 32 edges (two 16-row A tiles).
__global__ __launch_bounds__(64) void k_edge(
    const float* __restrict__ pos, const int* __restrict__ g,
    const float* __restrict__ eb0, const float* __restrict__ eb1,
    const float* __restrict__ eb2, const float* __restrict__ eg,
    const float* __restrict__ ebeta, const unsigned short* __restrict__ xbf,
    const unsigned short* __restrict__ wfrag, float* __restrict__ aggr) {
  constexpr int PITCH = 288;  // cols: 0..127 x_i | 128..255 x_j | 256..259 geo | pad
  __shared__ unsigned short act[2 * 32 * PITCH];
  __shared__ int sidxi[64];
  __shared__ int sidxj[64];
  const int tid = threadIdx.x;
  const int wave = tid >> 5, lane = tid & 31;
  const int half = lane >> 4, ln = lane & 15;
  unsigned short* wact = act + wave * 32 * PITCH;
  const int ebase = blockIdx.x * 64 + wave * 32;

  // indices + geometric features: one edge per lane (all 32 lanes)
  {
    int e = ebase + lane;
    int ok = (e < EE);
    int ec = ok ? e : 0;
    int i = g[ec], j = g[EE + ec];
    sidxi[wave * 32 + lane] = i;
    sidxj[wave * 32 + lane] = ok ? j : -1;
    float dx = pos[i * 3 + 0] - pos[j * 3 + 0];
    float dy = pos[i * 3 + 1] - pos[j * 3 + 1];
    float dz = pos[i * 3 + 2] - pos[j * 3 + 2];
    float nr = sqrtf(dx * dx + dy * dy + dz * dz);
    unsigned short* rp = wact + lane * PITCH + 256;
    uint4 z0;
    z0.x = (uint32_t)f2bf(dx) | ((uint32_t)f2bf(dy) << 16);
    z0.y = (uint32_t)f2bf(dz) | ((uint32_t)f2bf(nr) << 16);
    z0.z = 0u; z0.w = 0u;
    uint4 zz; zz.x = zz.y = zz.z = zz.w = 0u;
    *(uint4*)(rp + 0)  = z0;
    *(uint4*)(rp + 8)  = zz;
    *(uint4*)(rp + 16) = zz;
    *(uint4*)(rp + 24) = zz;
  }
  __syncthreads();

  // stage gathered x_i / x_j rows (bf16) into LDS
#pragma unroll 8
  for (int r = 0; r < 32; r++) {
    int jj = sidxj[wave * 32 + r]; if (jj < 0) jj = 0;
    int node = half ? jj : sidxi[wave * 32 + r];
    uint4 v = *(const uint4*)(xbf + (size_t)node * DD + ln * 8);
    *(uint4*)(wact + r * PITCH + half * 128 + ln * 8) = v;
  }
  __syncthreads();

  const unsigned short* arow0 = wact + ln * PITCH;
  const unsigned short* arow1 = wact + (16 + ln) * PITCH;
  v8f acc0[8], acc1[8];

  // layer 0: K=288
  init_acc2(acc0, acc1, eb0, ln);
  mlp_layer2<9>(acc0, acc1, arow0, arow1, 0, wfrag, FB_EW0, half, lane);
  __syncthreads();
  store_act_relu2(acc0, acc1, wact, PITCH, 0, half, ln);     // overwrite x_i half
  __syncthreads();

  // layer 1: read cols 0.., write cols 128..
  init_acc2(acc0, acc1, eb1, ln);
  mlp_layer2<4>(acc0, acc1, arow0, arow1, 0, wfrag, FB_EW1, half, lane);
  __syncthreads();
  store_act_relu2(acc0, acc1, wact, PITCH, 128, half, ln);   // overwrite x_j half
  __syncthreads();

  // layer 2: read cols 128.., no ReLU
  init_acc2(acc0, acc1, eb2, ln);
  mlp_layer2<4>(acc0, acc1, arow0, arow1, 128, wfrag, FB_EW2, half, lane);
  __syncthreads();

  // LayerNorm: dump f32 32x128 into LDS, one lane per edge row, two passes
  float* lnb = (float*)wact;
#pragma unroll
  for (int nt = 0; nt < 8; nt++) {
    int n = nt * 16 + ln;
#pragma unroll
    for (int r = 0; r < 8; r++) {
      lnb[(half * 8 + r) * DD + n]        = acc0[nt][r];
      lnb[(16 + half * 8 + r) * DD + n]   = acc1[nt][r];
    }
  }
  __syncthreads();

  const float* lr = lnb + lane * DD;
  float s = 0.0f, sq = 0.0f;
#pragma unroll
  for (int t = 0; t < 32; t++) {
    float4 q = ((const float4*)lr)[t];
    s  += q.x + q.y + q.z + q.w;
    sq += q.x * q.x + q.y * q.y + q.z * q.z + q.w * q.w;
  }
  float mean = s * (1.0f / 128.0f);
  float var  = sq * (1.0f / 128.0f) - mean * mean;
  float rs   = rsqrtf(var + 1e-5f);

  int j = sidxj[wave * 32 + lane];
  if (j >= 0) {
    float* ap = aggr + (size_t)j * DD;
#pragma unroll
    for (int t = 0; t < 32; t++) {
      float4 q  = ((const float4*)lr)[t];
      float4 gq = ((const float4*)eg)[t];
      float4 bq = ((const float4*)ebeta)[t];
      atomicAdd(ap + 4 * t + 0, (q.x - mean) * rs * gq.x + bq.x);
      atomicAdd(ap + 4 * t + 1, (q.y - mean) * rs * gq.y + bq.y);
      atomicAdd(ap + 4 * t + 2, (q.z - mean) * rs * gq.z + bq.z);
      atomicAdd(ap + 4 * t + 3, (q.w - mean) * rs * gq.w + bq.w);
    }
  }
}

// ---------------- node MLP + residual ----------------
__global__ __launch_bounds__(64) void k_node(
    const float* __restrict__ x, const float* __restrict__ nb0,
    const float* __restrict__ nb1, const float* __restrict__ nb2,
    const float* __restrict__ ng, const float* __restrict__ nbeta,
    const unsigned short* __restrict__ xbf, const float* __restrict__ aggr,
    const unsigned short* __restrict__ wfrag, float* __restrict__ out) {
  constexpr int PITCH = 256;  // cols: 0..127 x | 128..255 aggr
  __shared__ unsigned short act[2 * 32 * PITCH];
  const int tid = threadIdx.x;
  const int wave = tid >> 5, lane = tid & 31;
  const int half = lane >> 4, ln = lane & 15;
  unsigned short* wact = act + wave * 32 * PITCH;
  const int nbase = blockIdx.x * 64 + wave * 32;

#pragma unroll 8
  for (int r = 0; r < 32; r++) {
    int node = nbase + r; if (node >= NN) node = 0;
    if (half == 0) {
      uint4 v = *(const uint4*)(xbf + (size_t)node * DD + ln * 8);
      *(uint4*)(wact + r * PITCH + ln * 8) = v;
    } else {
      const float4* ar = (const float4*)(aggr + (size_t)node * DD + ln * 8);
      float4 a0 = ar[0], a1 = ar[1];
      uint4 v;
      v.x = (uint32_t)f2bf(a0.x) | ((uint32_t)f2bf(a0.y) << 16);
      v.y = (uint32_t)f2bf(a0.z) | ((uint32_t)f2bf(a0.w) << 16);
      v.z = (uint32_t)f2bf(a1.x) | ((uint32_t)f2bf(a1.y) << 16);
      v.w = (uint32_t)f2bf(a1.z) | ((uint32_t)f2bf(a1.w) << 16);
      *(uint4*)(wact + r * PITCH + 128 + ln * 8) = v;
    }
  }
  __syncthreads();

  const unsigned short* arow0 = wact + ln * PITCH;
  const unsigned short* arow1 = wact + (16 + ln) * PITCH;
  v8f acc0[8], acc1[8];

  init_acc2(acc0, acc1, nb0, ln);
  mlp_layer2<8>(acc0, acc1, arow0, arow1, 0, wfrag, FB_NW0, half, lane);
  __syncthreads();
  store_act_relu2(acc0, acc1, wact, PITCH, 0, half, ln);
  __syncthreads();

  init_acc2(acc0, acc1, nb1, ln);
  mlp_layer2<4>(acc0, acc1, arow0, arow1, 0, wfrag, FB_NW1, half, lane);
  __syncthreads();
  store_act_relu2(acc0, acc1, wact, PITCH, 128, half, ln);
  __syncthreads();

  init_acc2(acc0, acc1, nb2, ln);
  mlp_layer2<4>(acc0, acc1, arow0, arow1, 128, wfrag, FB_NW2, half, lane);
  __syncthreads();

  float* lnb = (float*)wact;
#pragma unroll
  for (int nt = 0; nt < 8; nt++) {
    int n = nt * 16 + ln;
#pragma unroll
    for (int r = 0; r < 8; r++) {
      lnb[(half * 8 + r) * DD + n]      = acc0[nt][r];
      lnb[(16 + half * 8 + r) * DD + n] = acc1[nt][r];
    }
  }
  __syncthreads();

  const float* lr = lnb + lane * DD;
  float s = 0.0f, sq = 0.0f;
#pragma unroll
  for (int t = 0; t < 32; t++) {
    float4 q = ((const float4*)lr)[t];
    s  += q.x + q.y + q.z + q.w;
    sq += q.x * q.x + q.y * q.y + q.z * q.z + q.w * q.w;
  }
  float mean = s * (1.0f / 128.0f);
  float var  = sq * (1.0f / 128.0f) - mean * mean;
  float rs   = rsqrtf(var + 1e-5f);

  int node = nbase + lane;
  if (node < NN) {
    const float4* xr = (const float4*)(x + (size_t)node * DD);
    float4* orow = (float4*)(out + (size_t)node * DD);
#pragma unroll
    for (int t = 0; t < 32; t++) {
      float4 q  = ((const float4*)lr)[t];
      float4 gq = ((const float4*)ng)[t];
      float4 bq = ((const float4*)nbeta)[t];
      float4 xq = xr[t];
      float4 o;
      o.x = (q.x - mean) * rs * gq.x + bq.x + xq.x;
      o.y = (q.y - mean) * rs * gq.y + bq.y + xq.y;
      o.z = (q.z - mean) * rs * gq.z + bq.z + xq.z;
      o.w = (q.w - mean) * rs * gq.w + bq.w + xq.w;
      orow[t] = o;
    }
  }
}

// ---------------- launch ----------------
extern "C" void kernel_launch(void* const* d_in, const int* in_sizes, int n_in,
                              void* d_out, int out_size, void* d_ws, size_t ws_size,
                              hipStream_t stream) {
  const float* x     = (const float*)d_in[0];
  const float* pos   = (const float*)d_in[1];
  const int*   g     = (const int*)d_in[2];
  const float* ew0   = (const float*)d_in[3];
  const float* eb0   = (const float*)d_in[4];
  const float* ew1   = (const float*)d_in[5];
  const float* eb1   = (const float*)d_in[6];
  const float* ew2   = (const float*)d_in[7];
  const float* eb2   = (const float*)d_in[8];
  const float* eg    = (const float*)d_in[9];
  const float* ebeta = (const float*)d_in[10];
  const float* nw0   = (const float*)d_in[11];
  const float* nb0   = (const float*)d_in[12];
  const float* nw1   = (const float*)d_in[13];
  const float* nb1   = (const float*)d_in[14];
  const float* nw2   = (const float*)d_in[15];
  const float* nb2   = (const float*)d_in[16];
  const float* ng    = (const float*)d_in[17];
  const float* nbeta = (const float*)d_in[18];

  char* ws = (char*)d_ws;
  float*          aggr  = (float*)(ws + OFF_AGGR);
  unsigned short* xbf   = (unsigned short*)(ws + OFF_XBF);
  unsigned short* wfrag = (unsigned short*)(ws + OFF_WFRAG);
  float* out = (float*)d_out;

  int nzero4 = NN * DD / 4;
  k_zero<<<(nzero4 + 255) / 256, 256, 0, stream>>>((uint4*)aggr, nzero4);
  int n4 = NN * DD / 4;
  k_xbf<<<(n4 + 255) / 256, 256, 0, stream>>>(x, xbf, n4);
  k_prep<<<(NFRAG * 512) / 256, 256, 0, stream>>>(ew0, ew1, ew2, nw0, nw1, nw2, wfrag);

  k_edge<<<(EE + 63) / 64, 64, 0, stream>>>(pos, g, eb0, eb1, eb2, eg, ebeta,
                                            xbf, wfrag, aggr);
  k_node<<<(NN + 63) / 64, 64, 0, stream>>>(x, nb0, nb1, nb2, ng, nbeta,
                                            xbf, aggr, wfrag, out);
}
